// Model_17274358464991
// MI455X (gfx1250) — matrix operand order repaired and checked by reading, not compiled, not run on hardware
//
#include <hip/hip_runtime.h>
#include <math.h>

#define CDIV(a,b) (((a)+(b)-1)/(b))

typedef _Float16 f16t;
typedef __attribute__((ext_vector_type(16))) _Float16 v16h;
typedef __attribute__((ext_vector_type(8)))  _Float16 v8h;
typedef __attribute__((ext_vector_type(8)))  float    v8f;

static constexpr int BB   = 16;
static constexpr int NN   = 207;
static constexpr int NP   = 224;  // node dim padded to 7*32 (K) / 14*16 (M)
static constexpr int CH   = 32;   // residual/conv channels
static constexpr int SCC  = 64;   // skip channels
static constexpr int ECC  = 128;  // end channels
static constexpr int PRED = 96;
static constexpr int TT0  = 336;
static constexpr int NDIM = 40;
static constexpr float ALPHA = 0.05f;

__device__ __forceinline__ float sigmoidf_(float x) { return 1.0f / (1.0f + expf(-x)); }

// ---------------------------------------------------------------------------
// xin[b,n,t] = input[b,t,n] * mask[b,t,n]
__global__ void xin_kernel(const float* __restrict__ in, const float* __restrict__ mk,
                           float* __restrict__ XIN) {
    long idx = (long)blockIdx.x * blockDim.x + threadIdx.x;
    long total = (long)BB * NN * TT0;
    if (idx >= total) return;
    int t = idx % TT0; long r = idx / TT0;
    int n = r % NN; int b = r / NN;
    long src = ((long)b * TT0 + t) * NN + n;
    XIN[idx] = in[src] * mk[src];
}

// ---------------------------------------------------------------------------
// build_graph: n1/n2 = tanh(3*(emb @ W + b))     [207,40]
__global__ void graph_n12_kernel(const float* __restrict__ e1, const float* __restrict__ e2,
                                 const float* __restrict__ W1, const float* __restrict__ b1,
                                 const float* __restrict__ W2, const float* __restrict__ b2,
                                 float* __restrict__ N1, float* __restrict__ N2) {
    int idx = blockIdx.x * blockDim.x + threadIdx.x;
    if (idx >= NN * NDIM) return;
    int j = idx % NDIM, i = idx / NDIM;
    float s1 = b1[j], s2 = b2[j];
    for (int k = 0; k < NDIM; ++k) {
        s1 += e1[i * NDIM + k] * W1[k * NDIM + j];
        s2 += e2[i * NDIM + k] * W2[k * NDIM + j];
    }
    N1[idx] = tanhf(3.0f * s1);
    N2[idx] = tanhf(3.0f * s2);
}

// a = relu(tanh(3*(n1 n2^T - n2 n1^T)))   [207,207]
__global__ void graph_a_kernel(const float* __restrict__ N1, const float* __restrict__ N2,
                               float* __restrict__ A) {
    int idx = blockIdx.x * blockDim.x + threadIdx.x;
    if (idx >= NN * NN) return;
    int j = idx % NN, i = idx / NN;
    float s = 0.f;
    for (int k = 0; k < NDIM; ++k)
        s += N1[i * NDIM + k] * N2[j * NDIM + k] - N2[i * NDIM + k] * N1[j * NDIM + k];
    float v = tanhf(3.0f * s);
    A[idx] = v > 0.f ? v : 0.f;
}

// per-row top-20 mask; adp = a * mask. Ties broken toward lower index (matches lax.top_k).
__global__ void topk_kernel(const float* __restrict__ A, float* __restrict__ ADP) {
    int row = blockIdx.x * blockDim.x + threadIdx.x;
    if (row >= NN) return;
    const float* ar = A + (long)row * NN;
    float* dr = ADP + (long)row * NN;
    bool used[NN];
    for (int w = 0; w < NN; ++w) { dr[w] = 0.f; used[w] = false; }
    for (int it = 0; it < 20; ++it) {
        int best = -1; float bv = -1.0e30f;
        for (int w = 0; w < NN; ++w)
            if (!used[w] && ar[w] > bv) { bv = ar[w]; best = w; }
        used[best] = true;
        dr[best] = ar[best];
    }
}

// A1 = rownorm(adp + I); A2 = rownorm(adp^T + I)
__global__ void adjnorm_kernel(const float* __restrict__ ADP,
                               float* __restrict__ A1, float* __restrict__ A2) {
    int v = blockIdx.x * blockDim.x + threadIdx.x;
    if (v >= NN) return;
    float s1 = 1.0f, s2 = 1.0f;
    for (int w = 0; w < NN; ++w) { s1 += ADP[(long)v * NN + w]; s2 += ADP[(long)w * NN + v]; }
    float r1 = 1.0f / s1, r2 = 1.0f / s2;
    for (int w = 0; w < NN; ++w) {
        float d = (w == v) ? 1.0f : 0.0f;
        A1[(long)v * NN + w] = (ADP[(long)v * NN + w] + d) * r1;
        A2[(long)v * NN + w] = (ADP[(long)w * NN + v] + d) * r2;
    }
}

// convert + zero-pad adjacency: A16[224][224] = f16(A[207][207]) padded with zeros
__global__ void a16_kernel(const float* __restrict__ A, _Float16* __restrict__ A16) {
    int idx = blockIdx.x * blockDim.x + threadIdx.x;
    if (idx >= NP * NP) return;
    int w = idx % NP, r = idx / NP;
    A16[idx] = (r < NN && w < NN) ? (f16t)A[(long)r * NN + w] : (f16t)0.0f;
}

// pack f32 [bc][207][T] -> f16 transposed padded [bc][Tp][224] (zero fill)
__global__ void pack16t_kernel(const float* __restrict__ X, _Float16* __restrict__ XT,
                               int T, int Tp) {
    long idx = (long)blockIdx.x * blockDim.x + threadIdx.x;
    long total = (long)BB * CH * Tp * NP;
    if (idx >= total) return;
    int w = idx % NP; long r = idx / NP;
    int t = r % Tp; long bc = r / Tp;
    XT[idx] = (w < NN && t < T) ? (f16t)X[(bc * NN + w) * T + t] : (f16t)0.0f;
}

// ---------------------------------------------------------------------------
// start conv (1x1, 1->32 channels)
__global__ void start_conv_kernel(const float* __restrict__ XIN, const float* __restrict__ w,
                                  const float* __restrict__ bias, float* __restrict__ X) {
    long idx = (long)blockIdx.x * blockDim.x + threadIdx.x;
    long total = (long)BB * CH * NN * TT0;
    if (idx >= total) return;
    int t = idx % TT0; long r = idx / TT0;
    int n = r % NN; r /= NN;
    int c = r % CH; int b = r / CH;
    X[idx] = XIN[((long)b * NN + n) * TT0 + t] * w[c] + bias[c];
}

// skip0: skip[b,co,n] = b[co] + sum_t xin[b,n,t] * w[co,t]
__global__ void skip0_kernel(const float* __restrict__ XIN, const float* __restrict__ W,
                             const float* __restrict__ bias, float* __restrict__ SKIP) {
    int idx = blockIdx.x * blockDim.x + threadIdx.x;
    if (idx >= BB * SCC * NN) return;
    int n = idx % NN; int r = idx / NN;
    int co = r % SCC; int b = r / SCC;
    const float* xp = XIN + ((long)b * NN + n) * TT0;
    const float* wp = W + (long)co * TT0;
    float s = bias[co];
    for (int t = 0; t < TT0; ++t) s += xp[t] * wp[t];
    SKIP[idx] = s;
}

// ---------------------------------------------------------------------------
// fused dilated-inception: xg = tanh(filter) * sigmoid(gate)
__global__ void inception_kernel(const float* __restrict__ X, float* __restrict__ XG,
    const float* fw0, const float* fw1, const float* fw2, const float* fw3,
    const float* fb0, const float* fb1, const float* fb2, const float* fb3,
    const float* gw0, const float* gw1, const float* gw2, const float* gw3,
    const float* gb0, const float* gb1, const float* gb2, const float* gb3,
    int Tin, int Tm, int dil) {
    long idx = (long)blockIdx.x * blockDim.x + threadIdx.x;
    long total = (long)BB * CH * NN * Tm;
    if (idx >= total) return;
    int t = idx % Tm; long r = idx / Tm;
    int n = r % NN; r /= NN;
    int oc = r % CH; int b = r / CH;
    int br = oc >> 3, c = oc & 7;
    int kk = (br == 0) ? 2 : (br == 1) ? 3 : (br == 2) ? 6 : 7;
    const float* fw = (br == 0) ? fw0 : (br == 1) ? fw1 : (br == 2) ? fw2 : fw3;
    const float* fb = (br == 0) ? fb0 : (br == 1) ? fb1 : (br == 2) ? fb2 : fb3;
    const float* gw = (br == 0) ? gw0 : (br == 1) ? gw1 : (br == 2) ? gw2 : gw3;
    const float* gb = (br == 0) ? gb0 : (br == 1) ? gb1 : (br == 2) ? gb2 : gb3;
    int base_t = t + dil * (7 - kk);
    float f = fb[c], g = gb[c];
    for (int ic = 0; ic < CH; ++ic) {
        const float* xp  = X + (((long)b * CH + ic) * NN + n) * Tin + base_t;
        const float* fwp = fw + ((long)c * CH + ic) * kk;
        const float* gwp = gw + ((long)c * CH + ic) * kk;
        for (int j = 0; j < kk; ++j) {
            float xv = xp[j * dil];
            f += xv * fwp[j];
            g += xv * gwp[j];
        }
    }
    XG[idx] = tanhf(f) * sigmoidf_(g);
}

// ---------------------------------------------------------------------------
// WMMA graph propagation:  res = alpha*Xs + (1-alpha) * (A x H)
//   A16 : [224][224] f16, zero-padded normalized adjacency (rows = v, cols = w)
//   Hin : [BC][Tp][224] f16 transposed padded activations (B operand)
//   Xs  : [BC][Tp][224] f16 transposed padded original x (for alpha-mix)
//   OUT : [BC][207][T] f32 (guarded store, feeds 1x1 conv accumulation)
//   O16 : [BC][Tp][224] f16 transposed padded (feeds the next propagation)
// One wave per 16x16 tile; K = node dim, 7 unguarded chunks of 32, fully unrolled.
__global__ __launch_bounds__(32) void prop_wmma_kernel(
    const _Float16* __restrict__ A16, const _Float16* __restrict__ Hin,
    const _Float16* __restrict__ Xs, float* __restrict__ OUT,
    _Float16* __restrict__ O16, int T, int Tp) {
    const int lane = threadIdx.x;
    const int t0 = blockIdx.x * 16;
    const int v0 = blockIdx.y * 16;
    const long bc = blockIdx.z;
    const long fs = (long)Tp * NP;
    const _Float16* h  = Hin + bc * fs;
    const _Float16* xs = Xs  + bc * fs;
    _Float16* o16 = O16 + bc * fs;
    float* out = OUT + bc * (long)NN * T;

    const int lo = lane & 15;
    const int hi = lane >> 4;
    const int t = t0 + lo;                        // column of B/D held by this lane
    const _Float16* arow = A16 + (long)(v0 + lo) * NP;  // A row for this lane
    const _Float16* brow = h + (long)t * NP;

    v8f acc = {};
    #pragma unroll
    for (int kb = 0; kb < 7; ++kb) {
        const int wb = kb * 32;
        // A fragment: lane holds row v0+lo; K sub-blocks {0-7,16-23} (lanes 0-15)
        // or {8-15,24-31} (lanes 16-31). Two 16B contiguous loads.
        v8h a0 = *(const v8h*)(arow + wb + hi * 8);
        v8h a1 = *(const v8h*)(arow + wb + 16 + hi * 8);
        // B fragment: lane holds column t; K sub-blocks {0-7,16-23} (lanes 0-15) or {8-15,24-31}. Two 16B loads.
        v8h b0 = *(const v8h*)(brow + wb + hi * 8);
        v8h b1 = *(const v8h*)(brow + wb + 16 + hi * 8);
        v16h av = __builtin_shufflevector(a0, a1, 0,1,2,3,4,5,6,7,8,9,10,11,12,13,14,15);
        v16h bv = __builtin_shufflevector(b0, b1, 0,1,2,3,4,5,6,7,8,9,10,11,12,13,14,15);
        acc = __builtin_amdgcn_wmma_f32_16x16x32_f16(
            false, av, false, bv, (short)0, acc, false, false);
    }
    // D: lane holds rows v0 + hi*8 + vg, column t. Contiguous in the transposed layout.
    v8h xv = *(const v8h*)(xs + (long)t * NP + v0 + hi * 8);
    v8h o;
    #pragma unroll
    for (int vg = 0; vg < 8; ++vg) {
        float res = ALPHA * (float)xv[vg] + (1.0f - ALPHA) * acc[vg];
        o[vg] = (f16t)res;
        int v = v0 + hi * 8 + vg;
        if (v < NN && t < T) out[(long)v * T + t] = res;
    }
    *(v8h*)(o16 + (long)t * NP + v0 + hi * 8) = o;   // padded region stays zero (A/B padded)
}

// ---------------------------------------------------------------------------
// acc init with both mixprop biases
__global__ void acc_init_kernel(float* __restrict__ ACC, const float* __restrict__ b1,
                                const float* __restrict__ b2, int T) {
    long idx = (long)blockIdx.x * blockDim.x + threadIdx.x;
    long total = (long)BB * CH * NN * T;
    if (idx >= total) return;
    int c = (idx / ((long)NN * T)) % CH;
    ACC[idx] = b1[c] + b2[c];
}

// acc += W[:, off:off+32] . Hc   (1x1 conv chunk, W is [32][96])
__global__ void conv1x1_acc_kernel(float* __restrict__ ACC, const float* __restrict__ Hc,
                                   const float* __restrict__ W, int off, int T) {
    long idx = (long)blockIdx.x * blockDim.x + threadIdx.x;
    long total = (long)BB * CH * NN * T;
    if (idx >= total) return;
    int t = idx % T; long r = idx / T;
    int n = r % NN; r /= NN;
    int co = r % CH; int b = r / CH;
    const long cs = (long)NN * T;
    const float* hp = Hc + ((long)b * CH) * cs + (long)n * T + t;
    const float* wr = W + (long)co * (3 * CH) + off;
    float s = 0.f;
    #pragma unroll 8
    for (int ci = 0; ci < CH; ++ci) s += wr[ci] * hp[(long)ci * cs];
    ACC[idx] += s;
}

// acc += residual tail: X has width Tin, take last Tm (offset = Tin - Tm)
__global__ void residual_add_kernel(float* __restrict__ ACC, const float* __restrict__ X,
                                    int Tin, int Tm) {
    long idx = (long)blockIdx.x * blockDim.x + threadIdx.x;
    long total = (long)BB * CH * NN * Tm;
    if (idx >= total) return;
    int t = idx % Tm; long r = idx / Tm;
    int n = r % NN; r /= NN;
    int c = r % CH; int b = r / CH;
    ACC[idx] += X[(((long)b * CH + c) * NN + n) * Tin + t + (Tin - Tm)];
}

// skip conv (full-width reduction): SKIP[b,co,n] += bias[co] + sum_{c,t} Xg*W
__global__ void skip_conv_kernel(const float* __restrict__ Xg, const float* __restrict__ W,
                                 const float* __restrict__ bias, float* __restrict__ SKIP,
                                 int T) {
    int idx = blockIdx.x * blockDim.x + threadIdx.x;
    if (idx >= BB * SCC * NN) return;
    int n = idx % NN; int r = idx / NN;
    int co = r % SCC; int b = r / SCC;
    float s = bias[co];
    for (int c = 0; c < CH; ++c) {
        const float* xp = Xg + (((long)b * CH + c) * NN + n) * T;
        const float* wp = W + ((long)co * CH + c) * T;
        for (int t = 0; t < T; ++t) s += xp[t] * wp[t];
    }
    SKIP[idx] += s;
}

// per-batch layernorm over (C,N,T) with elementwise affine; writes new X (width T)
__global__ void layernorm_kernel(const float* __restrict__ ACC, float* __restrict__ X,
                                 const float* __restrict__ lw, const float* __restrict__ lb,
                                 int T) {
    __shared__ float s1[256], s2[256];
    int b = blockIdx.x, tid = threadIdx.x;
    long cnt = (long)CH * NN * T;
    const float* src = ACC + (long)b * cnt;
    float a = 0.f, q = 0.f;
    for (long i = tid; i < cnt; i += 256) { float v = src[i]; a += v; q += v * v; }
    s1[tid] = a; s2[tid] = q;
    __syncthreads();
    for (int st = 128; st > 0; st >>= 1) {
        if (tid < st) { s1[tid] += s1[tid + st]; s2[tid] += s2[tid + st]; }
        __syncthreads();
    }
    float mu  = s1[0] / (float)cnt;
    float var = s2[0] / (float)cnt - mu * mu;
    float inv = rsqrtf(var + 1e-5f);
    float* dst = X + (long)b * cnt;
    for (long i = tid; i < cnt; i += 256)
        dst[i] = (src[i] - mu) * inv * lw[i] + lb[i];
}

// end head: e1 = relu(W1 . relu(skip) + b1);  out = W2 . e1 + b2
__global__ void end1_kernel(const float* __restrict__ SKIP, const float* __restrict__ W1,
                            const float* __restrict__ b1, float* __restrict__ E1) {
    int idx = blockIdx.x * blockDim.x + threadIdx.x;
    if (idx >= BB * ECC * NN) return;
    int n = idx % NN; int r = idx / NN;
    int k = r % ECC; int b = r / ECC;
    float s = b1[k];
    for (int c = 0; c < SCC; ++c) {
        float sv = SKIP[((long)b * SCC + c) * NN + n];
        sv = sv > 0.f ? sv : 0.f;
        s += sv * W1[(long)k * SCC + c];
    }
    E1[idx] = s > 0.f ? s : 0.f;
}

__global__ void end2_kernel(const float* __restrict__ E1, const float* __restrict__ W2,
                            const float* __restrict__ b2, float* __restrict__ OUT) {
    int idx = blockIdx.x * blockDim.x + threadIdx.x;
    if (idx >= BB * PRED * NN) return;
    int n = idx % NN; int r = idx / NN;
    int p = r % PRED; int b = r / PRED;
    float s = b2[p];
    for (int k = 0; k < ECC; ++k)
        s += E1[((long)b * ECC + k) * NN + n] * W2[(long)p * ECC + k];
    OUT[idx] = s;
}

// ---------------------------------------------------------------------------
extern "C" void kernel_launch(void* const* d_in, const int* in_sizes, int n_in,
                              void* d_out, int out_size, void* d_ws, size_t ws_size,
                              hipStream_t stream) {
    (void)in_sizes; (void)n_in; (void)out_size; (void)ws_size;
    auto P = [&](int i) { return (const float*)d_in[i]; };

    const float* in_    = P(0);
    const float* mask_  = P(1);
    const float* emb1   = P(2);
    const float* emb2   = P(3);
    const float* gl1w   = P(4);
    const float* gl1b   = P(5);
    const float* gl2w   = P(6);
    const float* gl2b   = P(7);
    const float* startw = P(8);
    const float* startb = P(9);
    const float* skip0w = P(10);
    const float* skip0b = P(11);
    const float* skipEw = P(12);
    const float* skipEb = P(13);
    const float* end1w  = P(14);
    const float* end1b  = P(15);
    const float* end2w  = P(16);
    const float* end2b  = P(17);

    // workspace carve-up (byte based, 128B aligned per buffer)
    char* base = (char*)d_ws;
    size_t off = 0;
    auto allocB = [&](size_t bytes) {
        void* p = base + off;
        off += (bytes + 127) & ~(size_t)127;
        return p;
    };
    const size_t WMAX = 330, TPMAX = 336;
    float* XIN  = (float*)allocB((size_t)BB * NN * TT0 * 4);
    float* X    = (float*)allocB((size_t)BB * CH * NN * TT0 * 4);
    float* XG   = (float*)allocB((size_t)BB * CH * NN * WMAX * 4);
    float* HA   = (float*)allocB((size_t)BB * CH * NN * WMAX * 4);
    float* HB   = (float*)allocB((size_t)BB * CH * NN * WMAX * 4);
    float* ACC  = (float*)allocB((size_t)BB * CH * NN * WMAX * 4);
    float* SKIP = (float*)allocB((size_t)BB * SCC * NN * 4);
    float* N1   = (float*)allocB((size_t)NN * NDIM * 4);
    float* N2   = (float*)allocB((size_t)NN * NDIM * 4);
    float* AMAT = (float*)allocB((size_t)NN * NN * 4);
    float* ADP  = (float*)allocB((size_t)NN * NN * 4);
    float* A1   = (float*)allocB((size_t)NN * NN * 4);
    float* A2   = (float*)allocB((size_t)NN * NN * 4);
    float* E1   = (float*)allocB((size_t)BB * ECC * NN * 4);
    _Float16* A16_1 = (_Float16*)allocB((size_t)NP * NP * 2);
    _Float16* A16_2 = (_Float16*)allocB((size_t)NP * NP * 2);
    _Float16* XG16  = (_Float16*)allocB((size_t)BB * CH * TPMAX * NP * 2);
    _Float16* HA16  = (_Float16*)allocB((size_t)BB * CH * TPMAX * NP * 2);
    _Float16* HB16  = (_Float16*)allocB((size_t)BB * CH * TPMAX * NP * 2);

    const int BLK = 256;

    // 1) masked transpose input
    {
        long total = (long)BB * NN * TT0;
        xin_kernel<<<CDIV(total, BLK), BLK, 0, stream>>>(in_, mask_, XIN);
    }
    // 2) graph construction
    graph_n12_kernel<<<CDIV(NN * NDIM, BLK), BLK, 0, stream>>>(emb1, emb2, gl1w, gl1b, gl2w, gl2b, N1, N2);
    graph_a_kernel<<<CDIV(NN * NN, BLK), BLK, 0, stream>>>(N1, N2, AMAT);
    topk_kernel<<<CDIV(NN, BLK), BLK, 0, stream>>>(AMAT, ADP);
    adjnorm_kernel<<<CDIV(NN, BLK), BLK, 0, stream>>>(ADP, A1, A2);
    a16_kernel<<<CDIV(NP * NP, BLK), BLK, 0, stream>>>(A1, A16_1);
    a16_kernel<<<CDIV(NP * NP, BLK), BLK, 0, stream>>>(A2, A16_2);
    // 3) start conv + skip0
    {
        long total = (long)BB * CH * NN * TT0;
        start_conv_kernel<<<CDIV(total, BLK), BLK, 0, stream>>>(XIN, startw, startb, X);
    }
    skip0_kernel<<<CDIV(BB * SCC * NN, BLK), BLK, 0, stream>>>(XIN, skip0w, skip0b, SKIP);

    // 4) layer loop
    const int TinA[3] = {336, 330, 318};
    const int TmA[3]  = {330, 318, 294};
    const int TpA[3]  = {336, 320, 304};   // Tm rounded up to 16
    const int dilA[3] = {1, 2, 4};
    for (int L = 0; L < 3; ++L) {
        const int base_i = 18 + L * 24;
        const int Tin = TinA[L], Tm = TmA[L], Tp = TpA[L], dil = dilA[L];
        const float *fw0 = P(base_i + 0), *fw1 = P(base_i + 1), *fw2 = P(base_i + 2), *fw3 = P(base_i + 3);
        const float *fb0 = P(base_i + 4), *fb1 = P(base_i + 5), *fb2 = P(base_i + 6), *fb3 = P(base_i + 7);
        const float *gw0 = P(base_i + 8), *gw1 = P(base_i + 9), *gw2 = P(base_i + 10), *gw3 = P(base_i + 11);
        const float *gb0 = P(base_i + 12), *gb1 = P(base_i + 13), *gb2 = P(base_i + 14), *gb3 = P(base_i + 15);
        const float *skw = P(base_i + 16), *skb = P(base_i + 17);
        const float *g1w = P(base_i + 18), *g1b = P(base_i + 19);
        const float *g2w = P(base_i + 20), *g2b = P(base_i + 21);
        const float *lnw = P(base_i + 22), *lnb = P(base_i + 23);

        long totTm = (long)BB * CH * NN * Tm;
        long totTp = (long)BB * CH * Tp * NP;
        // gated dilated inception
        inception_kernel<<<CDIV(totTm, BLK), BLK, 0, stream>>>(X, XG,
            fw0, fw1, fw2, fw3, fb0, fb1, fb2, fb3,
            gw0, gw1, gw2, gw3, gb0, gb1, gb2, gb3, Tin, Tm, dil);
        // skip accumulation from gated output
        skip_conv_kernel<<<CDIV(BB * SCC * NN, BLK), BLK, 0, stream>>>(XG, skw, skb, SKIP, Tm);
        // f16 transposed padded copy of xg for WMMA
        pack16t_kernel<<<CDIV(totTp, BLK), BLK, 0, stream>>>(XG, XG16, Tm, Tp);

        // mixprop(x, A1, g1) + mixprop(x, A2, g2), accumulated into ACC
        acc_init_kernel<<<CDIV(totTm, BLK), BLK, 0, stream>>>(ACC, g1b, g2b, Tm);
        dim3 pgrid(Tp / 16, NP / 16, BB * CH);
        const _Float16* Amats[2] = {A16_1, A16_2};
        const float* Wmats[2] = {g1w, g2w};
        for (int m = 0; m < 2; ++m) {
            const _Float16* Am = Amats[m];
            const float* Wm = Wmats[m];
            // chunk 0: x itself
            conv1x1_acc_kernel<<<CDIV(totTm, BLK), BLK, 0, stream>>>(ACC, XG, Wm, 0, Tm);
            // h1 = a*x + (1-a) A.h0    (WMMA)
            prop_wmma_kernel<<<pgrid, 32, 0, stream>>>(Am, XG16, XG16, HA, HA16, Tm, Tp);
            conv1x1_acc_kernel<<<CDIV(totTm, BLK), BLK, 0, stream>>>(ACC, HA, Wm, 32, Tm);
            // h2 = a*x + (1-a) A.h1    (WMMA)
            prop_wmma_kernel<<<pgrid, 32, 0, stream>>>(Am, HA16, XG16, HB, HB16, Tm, Tp);
            conv1x1_acc_kernel<<<CDIV(totTm, BLK), BLK, 0, stream>>>(ACC, HB, Wm, 64, Tm);
        }
        // + residual tail, then layernorm -> new X (width Tm)
        residual_add_kernel<<<CDIV(totTm, BLK), BLK, 0, stream>>>(ACC, X, Tin, Tm);
        layernorm_kernel<<<BB, 256, 0, stream>>>(ACC, X, lnw, lnb, Tm);
    }

    // 5) final skip + head
    skip_conv_kernel<<<CDIV(BB * SCC * NN, BLK), BLK, 0, stream>>>(X, skipEw, skipEb, SKIP, 294);
    end1_kernel<<<CDIV(BB * ECC * NN, BLK), BLK, 0, stream>>>(SKIP, end1w, end1b, E1);
    end2_kernel<<<CDIV(BB * PRED * NN, BLK), BLK, 0, stream>>>(E1, end2w, end2b, (float*)d_out);
}
